// BlockSelfAttention_2542620639911
// MI455X (gfx1250) — compile-verified, matrix operand order checked by reading, not run on hardware
//
#include <hip/hip_runtime.h>
#include <math.h>

typedef __attribute__((ext_vector_type(16))) __bf16 v16bf;
typedef __attribute__((ext_vector_type(8)))  float  v8f;
typedef __attribute__((ext_vector_type(2)))  __bf16 v2bf;

#define CCH   128   // channels
#define LTOK  64    // tokens per window (8x8)
#define XS    136   // bf16 row stride for [64][C] buffers (272B/row -> conflict-free, 16B aligned)
#define VS    72    // bf16 row stride for vT [128][64] (144B/row)
#define AS    68    // bf16 row stride for attn [4*64][64] (136B/row, fits the overlay exactly)

#define WQKV_ELEMS (384 * 128)
#define WO_ELEMS   (128 * 128)

union BF16Tile { v16bf v; uint4 q[2]; unsigned int u[8]; };

__device__ __forceinline__ unsigned short f2bf_u(float f) {
  return __builtin_bit_cast(unsigned short, (__bf16)f);   // native v_cvt
}

#if __has_builtin(__builtin_amdgcn_cvt_pk_bf16_f32)
__device__ __forceinline__ unsigned int pack2(float lo, float hi) {
  v2bf r = __builtin_amdgcn_cvt_pk_bf16_f32(lo, hi);      // single v_cvt_pk_bf16_f32
  return __builtin_bit_cast(unsigned int, r);
}
#else
__device__ __forceinline__ unsigned int pack2(float lo, float hi) {
  return (unsigned int)f2bf_u(lo) | ((unsigned int)f2bf_u(hi) << 16);
}
#endif

// 16x32 bf16 tile in WMMA A/B register layout from a row-major bf16 LDS array.
// Lane layout: row = lane%16, element pair p -> k = (p<4?0:16)+(p&3)*2 + 8*(lane/16).
// Pairs u[0..3] = k base+0..7 (contiguous), u[4..7] = k base+16..23 (contiguous):
// exactly two ds_load_b128 per lane.
__device__ __forceinline__ v16bf lds_tile16x32(const unsigned short* base,
                                               int stride, int r0, int k0, int lane) {
  const unsigned short* row =
      base + (r0 + (lane & 15)) * stride + k0 + ((lane >> 4) & 1) * 8;
  BF16Tile t;
  t.q[0] = *(const uint4*)(row);        // k offsets 0..7
  t.q[1] = *(const uint4*)(row + 16);   // k offsets 16..23
  return t.v;
}

// 32x16 B tile from row-major *bf16* weights W[D][K] (pre-converted in d_ws):
// B[k][n] = W[n0+n][k0+k]. Two global_load_b128 per lane, no conversion VALU.
__device__ __forceinline__ v16bf gmem_wtile_bf(const unsigned short* __restrict__ W,
                                               int ld, int n0, int k0, int lane) {
  const unsigned short* row =
      W + (size_t)(n0 + (lane & 15)) * ld + k0 + ((lane >> 4) & 1) * 8;
  BF16Tile t;
  t.q[0] = *(const uint4*)(row);
  t.q[1] = *(const uint4*)(row + 16);
  return t.v;
}

#define WMMA_BF16(a, b, c) \
  __builtin_amdgcn_wmma_f32_16x16x32_bf16(false, (a), false, (b), (short)0, (c), false, false)

// ---------------- Prologue: fp32 -> bf16 weight conversion into d_ws ---------
__global__ __launch_bounds__(256) void convert_weights_kernel(
    const float* __restrict__ wqkv, const float* __restrict__ wo,
    unsigned short* __restrict__ wbf) {
  const int i = blockIdx.x * blockDim.x + threadIdx.x;     // one float4 each
  const int qn = WQKV_ELEMS / 4;
  const int total = (WQKV_ELEMS + WO_ELEMS) / 4;
  if (i >= total) return;
  float4 f = (i < qn) ? ((const float4*)wqkv)[i] : ((const float4*)wo)[i - qn];
  unsigned int* dst = (unsigned int*)wbf + (size_t)i * 2;
  dst[0] = pack2(f.x, f.y);
  dst[1] = pack2(f.z, f.w);
}

__global__ __launch_bounds__(256) void swin_window_attn_kernel(
    const float* __restrict__ x, const float* __restrict__ ln_w,
    const float* __restrict__ ln_b, const unsigned short* __restrict__ wbf,
    const float* __restrict__ bqkv, const float* __restrict__ bo,
    const float* __restrict__ gamma, float* __restrict__ out) {
  const unsigned short* wqkv_bf = wbf;                 // [384][128] bf16
  const unsigned short* wo_bf   = wbf + WQKV_ELEMS;    // [128][128] bf16

  // bufA: [ s_x | s_q ] during phases 1-3a; overlaid by attn (4 heads x 64 x AS)
  __shared__ unsigned short s_bufA[2 * LTOK * XS];
  __shared__ unsigned short s_k[LTOK * XS];
  __shared__ unsigned short s_vT[CCH * VS];
  __shared__ unsigned short s_o[LTOK * XS];

  unsigned short* s_x    = s_bufA;
  unsigned short* s_q    = s_bufA + LTOK * XS;
  unsigned short* s_attn = s_bufA;               // overlay after barrier 3

  const int tid  = threadIdx.x;
  const int lane = tid & 31;
  const int wave = tid >> 5;
  const int half = (lane >> 4) & 1;

  const int win = blockIdx.x;        // one window per workgroup
  const int b   = win >> 10;
  const int wh  = (win >> 5) & 31;
  const int ww  = win & 31;
  const int y0  = wh * 8, x0 = ww * 8;

  // ---------------- Phase 1: LayerNorm -> s_x (bf16) ----------------
  {
    const int tok = tid >> 2;                    // 4 threads per token
    const int c0  = (tid & 3) * 32;
    const int yy = y0 + (tok >> 3);
    const int xx = x0 + (tok & 7);
    const float* px = x + ((size_t)(b * 256 + yy) * 256 + xx) * CCH + c0;
    float4 vbuf[8];
    float s = 0.f, ss = 0.f;
#pragma unroll
    for (int i = 0; i < 8; ++i) {
      float4 v = ((const float4*)px)[i];
      vbuf[i] = v;
      s  += v.x + v.y + v.z + v.w;
      ss += v.x * v.x + v.y * v.y + v.z * v.z + v.w * v.w;
    }
    s  += __shfl_xor(s, 1);  s  += __shfl_xor(s, 2);
    ss += __shfl_xor(ss, 1); ss += __shfl_xor(ss, 2);
    const float mean = s * (1.f / CCH);
    const float var  = ss * (1.f / CCH) - mean * mean;
    const float rstd = rsqrtf(var + 1e-5f);
    unsigned int* dst = (unsigned int*)(s_x + tok * XS + c0);
#pragma unroll
    for (int i = 0; i < 8; ++i) {
      float4 v = vbuf[i];
      float4 gw = ((const float4*)(ln_w + c0))[i];
      float4 gb = ((const float4*)(ln_b + c0))[i];
      dst[i * 2 + 0] = pack2((v.x - mean) * rstd * gw.x + gb.x,
                             (v.y - mean) * rstd * gw.y + gb.y);
      dst[i * 2 + 1] = pack2((v.z - mean) * rstd * gw.z + gb.z,
                             (v.w - mean) * rstd * gw.w + gb.w);
    }
  }
  __syncthreads();

  // ------- Phase 2: fused QKV projection, single pass over A-tiles -------
  // Wave w owns output columns [16w,16w+16) of each of Q, K, V.
  // 12 weight B-tiles cached in VGPRs; each A-tile feeds 3 WMMAs (Q,K,V).
  const float qscale = 0.17677669529663687f;     // 1/sqrt(DIM_HEAD)
  {
    v16bf Bq[4], Bk4[4], Bv4[4];
#pragma unroll
    for (int kt = 0; kt < 4; ++kt) {
      Bq[kt]  = gmem_wtile_bf(wqkv_bf, CCH, wave * 16,        kt * 32, lane);
      Bk4[kt] = gmem_wtile_bf(wqkv_bf, CCH, (wave + 8) * 16,  kt * 32, lane);
      Bv4[kt] = gmem_wtile_bf(wqkv_bf, CCH, (wave + 16) * 16, kt * 32, lane);
    }
    const int ng = wave * 16 + (lane & 15);
    const float biasq = bqkv[ng];
    const float biask = bqkv[128 + ng];
    const float biasv = bqkv[256 + ng];
#pragma unroll
    for (int mT = 0; mT < 4; ++mT) {
      v8f aq = {}, ak = {}, av = {};
#pragma unroll
      for (int kt = 0; kt < 4; ++kt) {
        v16bf At = lds_tile16x32(s_x, XS, mT * 16, kt * 32, lane);
        aq = WMMA_BF16(At, Bq[kt],  aq);
        ak = WMMA_BF16(At, Bk4[kt], ak);
        av = WMMA_BF16(At, Bv4[kt], av);
      }
#pragma unroll
      for (int r = 0; r < 8; ++r) {
        const int m = mT * 16 + r + half * 8;
        s_q[m * XS + ng]  = f2bf_u((aq[r] + biasq) * qscale);
        s_k[m * XS + ng]  = f2bf_u(ak[r] + biask);
        s_vT[ng * VS + m] = f2bf_u(av[r] + biasv);   // V stored transposed
      }
    }
  }
  __syncthreads();

  // ---------------- Phase 3a: scores = q . k^T (per head, 2 waves/head) ----
  const int head  = wave >> 1;
  const int whalf = wave & 1;                    // which 32 query rows
  v8f S[2][4];
  {
    v16bf Bk[4];
#pragma unroll
    for (int nT = 0; nT < 4; ++nT)
      Bk[nT] = lds_tile16x32(s_k, XS, nT * 16, head * 32, lane);
#pragma unroll
    for (int mT = 0; mT < 2; ++mT) {
      const int mAbs = whalf * 32 + mT * 16;
      v16bf Aq = lds_tile16x32(s_q, XS, mAbs, head * 32, lane);
#pragma unroll
      for (int nT = 0; nT < 4; ++nT) {
        v8f acc = {};
        S[mT][nT] = WMMA_BF16(Aq, Bk[nT], acc);
      }
    }
  }
  __syncthreads();  // all q/k reads done -> bufA reusable as attn

  // ---------------- Phase 3b: softmax rows, write attn bf16 ----------------
#pragma unroll
  for (int mT = 0; mT < 2; ++mT) {
#pragma unroll
    for (int r = 0; r < 8; ++r) {
      float mx = -3.0e38f;
#pragma unroll
      for (int nT = 0; nT < 4; ++nT) mx = fmaxf(mx, S[mT][nT][r]);
      mx = fmaxf(mx, __shfl_xor(mx, 1));
      mx = fmaxf(mx, __shfl_xor(mx, 2));
      mx = fmaxf(mx, __shfl_xor(mx, 4));
      mx = fmaxf(mx, __shfl_xor(mx, 8));         // reduce over n within 16-lane half
      float p[4], sum = 0.f;
#pragma unroll
      for (int nT = 0; nT < 4; ++nT) { p[nT] = __expf(S[mT][nT][r] - mx); sum += p[nT]; }
      sum += __shfl_xor(sum, 1);
      sum += __shfl_xor(sum, 2);
      sum += __shfl_xor(sum, 4);
      sum += __shfl_xor(sum, 8);
      const float inv = 1.f / sum;
      const int m = whalf * 32 + mT * 16 + r + half * 8;
      unsigned short* arow = s_attn + (head * LTOK + m) * AS + (lane & 15);
#pragma unroll
      for (int nT = 0; nT < 4; ++nT)
        arow[nT * 16] = f2bf_u(p[nT] * inv);
    }
  }
  // no barrier: each wave consumes only the attn rows it just wrote

  // ---------------- Phase 4: out = attn . v ----------------
  {
    v16bf Bv[2][2];
#pragma unroll
    for (int nT = 0; nT < 2; ++nT)
#pragma unroll
      for (int kt = 0; kt < 2; ++kt)
        Bv[nT][kt] = lds_tile16x32(s_vT, VS, head * 32 + nT * 16, kt * 32, lane);
#pragma unroll
    for (int mT = 0; mT < 2; ++mT) {
      const int mAbs = whalf * 32 + mT * 16;
      v16bf Aa[2];
#pragma unroll
      for (int kt = 0; kt < 2; ++kt)
        Aa[kt] = lds_tile16x32(s_attn + head * LTOK * AS, AS, mAbs, kt * 32, lane);
#pragma unroll
      for (int nT = 0; nT < 2; ++nT) {
        v8f acc = {};
#pragma unroll
        for (int kt = 0; kt < 2; ++kt)
          acc = WMMA_BF16(Aa[kt], Bv[nT][kt], acc);
        const int nc = head * 32 + nT * 16 + (lane & 15);
#pragma unroll
        for (int r = 0; r < 8; ++r)
          s_o[(mAbs + r + half * 8) * XS + nc] = f2bf_u(acc[r]);
      }
    }
  }
  __syncthreads();

  // ------- Phase 5: out projection + bias + LayerScale + window reverse -------
  {
    const int ng = wave * 16 + (lane & 15);      // 8 waves x 16 = 128 channels
    const float bias = bo[ng];
    const float g = gamma[ng];
    v16bf Bt[4];
#pragma unroll
    for (int kt = 0; kt < 4; ++kt)
      Bt[kt] = gmem_wtile_bf(wo_bf, CCH, wave * 16, kt * 32, lane);
#pragma unroll
    for (int mT = 0; mT < 4; ++mT) {
      v8f acc = {};
#pragma unroll
      for (int kt = 0; kt < 4; ++kt) {
        v16bf At = lds_tile16x32(s_o, XS, mT * 16, kt * 32, lane);
        acc = WMMA_BF16(At, Bt[kt], acc);
      }
#pragma unroll
      for (int r = 0; r < 8; ++r) {
        const int tok = mT * 16 + r + half * 8;
        const int yy = y0 + (tok >> 3);
        const int xx = x0 + (tok & 7);
        out[((size_t)(b * 256 + yy) * 256 + xx) * CCH + ng] = (acc[r] + bias) * g;
      }
    }
  }
}

extern "C" void kernel_launch(void* const* d_in, const int* in_sizes, int n_in,
                              void* d_out, int out_size, void* d_ws, size_t ws_size,
                              hipStream_t stream) {
  const float* x    = (const float*)d_in[0];
  const float* lnw  = (const float*)d_in[1];
  const float* lnb  = (const float*)d_in[2];
  const float* wqkv = (const float*)d_in[3];
  const float* bqkv = (const float*)d_in[4];
  const float* wo   = (const float*)d_in[5];
  const float* bo   = (const float*)d_in[6];
  const float* g    = (const float*)d_in[7];
  float* out = (float*)d_out;
  unsigned short* wbf = (unsigned short*)d_ws;   // 131072 B of bf16 weights

  // Prologue: convert weights fp32 -> bf16 row-major into scratch (L2-hot).
  {
    const int total4 = (WQKV_ELEMS + WO_ELEMS) / 4;      // 16384 float4s
    dim3 grid((total4 + 255) / 256), block(256);
    hipLaunchKernelGGL(convert_weights_kernel, grid, block, 0, stream, wqkv, wo, wbf);
  }

  const int B = in_sizes[0] / (256 * 256 * 128);     // 8
  dim3 grid(B * 32 * 32);                            // one 8x8 window per block
  dim3 block(256);                                   // 8 waves (wave32)
  hipLaunchKernelGGL(swin_window_attn_kernel, grid, block, 0, stream,
                     x, lnw, lnb, wbf, bqkv, bo, g, out);
}
